// GroupedQueryAttention_4389456577373
// MI455X (gfx1250) — matrix operand order repaired and checked by reading, not compiled, not run on hardware
//
#include <hip/hip_runtime.h>

// ---------------------------------------------------------------------------
// GroupedQueryAttention on MI455X (gfx1250, wave32, WMMA + async LDS DMA)
// HIDDEN=1024, NUM_HEADS=16, NUM_KV_HEADS=4, HEAD_DIM=64, B=2, SQ=SKV=2048
// All matmuls: v_wmma_f32_16x16x32_bf16 (f32 accumulate); operands converted
// to bf16 once; double-buffered GLOBAL_LOAD_ASYNC_TO_LDS_B128 staging.
// Attention: S^T = K Q^T and O^T = V^T P^T so each lane owns one q column ->
// softmax state, rescale and normalization are entirely per-lane.
// ---------------------------------------------------------------------------

typedef __attribute__((ext_vector_type(16))) __bf16 v16bf;
typedef __attribute__((ext_vector_type(8)))  float  v8f;
typedef __attribute__((ext_vector_type(4)))  int    v4i;
typedef unsigned short ushort_t;

union Frag16 {
    v16bf v;
    ushort_t u[16];
};

// hardware f32 -> bf16 (RNE) via native cast
static __device__ __forceinline__ ushort_t f2bf(float f) {
    union { __bf16 h; ushort_t u; } c;
    c.h = (__bf16)f;
    return c.u;
}

// raw v_exp_f32 (flush-to-zero underflow is fine for softmax)
static __device__ __forceinline__ float fast_exp2(float x) {
#if __has_builtin(__builtin_amdgcn_exp2f)
    return __builtin_amdgcn_exp2f(x);
#else
    return exp2f(x);
#endif
}

static __device__ __forceinline__ v8f wmma_bf16(const Frag16& a, const Frag16& b, v8f c) {
    return __builtin_amdgcn_wmma_f32_16x16x32_bf16(
        false, a.v, false, b.v, (short)0, c, false, false);
}

// ----- CDNA5 async global->LDS copy (ASYNCcnt-tracked DMA), guarded ---------
#if __has_builtin(__builtin_amdgcn_global_load_async_to_lds_b128)
#define HAVE_ASYNC_LDS 1
#else
#define HAVE_ASYNC_LDS 0
#endif

typedef __attribute__((address_space(1))) v4i g_v4i;   // global int4*
typedef __attribute__((address_space(3))) v4i l_v4i;   // LDS int4*

static __device__ __forceinline__ g_v4i* to_gl(const void* p) {
    return (g_v4i*)(unsigned long long)p;
}
static __device__ __forceinline__ l_v4i* to_lds(void* p) {
    return (l_v4i*)(unsigned int)(unsigned long long)p;
}

// one 16-byte per-lane async copy (wave32: 512B per instruction)
static __device__ __forceinline__ void async_cp16(const void* g, void* l) {
#if HAVE_ASYNC_LDS
    __builtin_amdgcn_global_load_async_to_lds_b128(to_gl(g), to_lds(l), 0, 0);
#else
    *(uint4*)l = *(const uint4*)g;
#endif
}
static __device__ __forceinline__ void wait_async0() {
#if __has_builtin(__builtin_amdgcn_s_wait_asynccnt)
    __builtin_amdgcn_s_wait_asynccnt(0);
#elif HAVE_ASYNC_LDS
    asm volatile("s_wait_asynccnt 0x0" ::: "memory");
#endif
}

// ----- fragment builders (ISA 16-bit A 16x32 / B 32x16 layouts) -------------
// A frag: lane holds row M=l16; k = kbA + (i&7) (+16 for i>=8); two b128 loads
static __device__ __forceinline__ void frag_a_bf16(Frag16& f, const ushort_t* row, int kbA) {
    ((uint4*)f.u)[0] = *(const uint4*)(row + kbA);
    ((uint4*)f.u)[1] = *(const uint4*)(row + kbA + 16);
}
// B frag: lane holds col N=l16; k = kbB + (i&7) (+16 for i>=8); two b128 loads
static __device__ __forceinline__ void frag_b_bf16(Frag16& f, const ushort_t* p) {
    ((uint4*)f.u)[0] = ((const uint4*)p)[0];
    ((uint4*)f.u)[1] = ((const uint4*)p)[2];
}

// epilogue store helpers
static __device__ __forceinline__ void store_out(float* p, float v)    { *p = v; }
static __device__ __forceinline__ void store_out(ushort_t* p, float v) { *p = f2bf(v); }

// ---------------------------------------------------------------------------
// f32 -> bf16 elementwise (one-time operand conversion), 8 elems/thread
// ---------------------------------------------------------------------------
__global__ __launch_bounds__(256)
void cvt_f32_bf16(const float* __restrict__ in, ushort_t* __restrict__ out, int n) {
    const int i = (blockIdx.x * 256 + threadIdx.x) * 8;
    if (i < n) {
        const float4 a = ((const float4*)(in + i))[0];
        const float4 b = ((const float4*)(in + i))[1];
        Frag16 r;
        r.u[0] = f2bf(a.x); r.u[1] = f2bf(a.y); r.u[2] = f2bf(a.z); r.u[3] = f2bf(a.w);
        r.u[4] = f2bf(b.x); r.u[5] = f2bf(b.y); r.u[6] = f2bf(b.z); r.u[7] = f2bf(b.w);
        *(uint4*)(out + i) = ((uint4*)r.u)[0];
    }
}

// ---------------------------------------------------------------------------
// GEMM:  C[M,N] = (A_bf16[M,K] * W_bf16[N,K]^T + bias[N]) * alpha
// 256 thr = 8 waves (4M x 2N), block tile 128x128, K-slab 64 (2 WMMA k-steps),
// double-buffered async DMA; inner loop is ds_load_b128 + v_wmma only.
// ---------------------------------------------------------------------------
#define GT_K 64

template <typename OutT>
__global__ __launch_bounds__(256)
void gemm_bf16(const ushort_t* __restrict__ A, const ushort_t* __restrict__ W,
               const float* __restrict__ bias, OutT* __restrict__ C,
               int M, int N, int K, float alpha) {
    __shared__ ushort_t As[2][128][72];   // 144B row stride (16B aligned, 36-bank)
    __shared__ ushort_t Ws[2][128][72];

    const int tid  = threadIdx.x;
    const int lane = tid & 31;
    const int wave = tid >> 5;
    const int wm   = wave >> 1;
    const int wn   = wave & 1;
    const int half = lane >> 4;
    const int l16  = lane & 15;
    const int m0   = blockIdx.x * 128;
    const int n0   = blockIdx.y * 128;
    const int kbA  = half * 8;
    const int kbB  = half * 8;

    v8f acc[2][4];
#pragma unroll
    for (int mi = 0; mi < 2; ++mi)
#pragma unroll
        for (int nj = 0; nj < 4; ++nj)
#pragma unroll
            for (int r = 0; r < 8; ++r) acc[mi][nj][r] = 0.0f;

    // per-wave DMA: 16 rows per wave, 8x 16B chunks per row, 4 ops/matrix
    const int rr  = wave * 16;                   // + p*4 + lane>>3
    const int ccs = (lane & 7) * 8;              // 16B chunk (shorts)

    auto issue = [&](int k0, int bufi) {
#pragma unroll
        for (int p = 0; p < 4; ++p) {
            const int r = rr + p * 4 + (lane >> 3);
            async_cp16(A + (size_t)(m0 + r) * K + k0 + ccs, &As[bufi][r][ccs]);
            async_cp16(W + (size_t)(n0 + r) * K + k0 + ccs, &Ws[bufi][r][ccs]);
        }
    };

    const int nslab = K / GT_K;
    issue(0, 0);
    for (int it = 0; it < nslab; ++it) {
        wait_async0();
        __syncthreads();
        if (it + 1 < nslab) issue((it + 1) * GT_K, (it + 1) & 1);
        const int cb = it & 1;

#pragma unroll
        for (int ks = 0; ks < 2; ++ks) {         // two 32-deep k-steps
            Frag16 af[2], bf[4];
#pragma unroll
            for (int mi = 0; mi < 2; ++mi)
                frag_a_bf16(af[mi], &As[cb][wm * 32 + mi * 16 + l16][ks * 32], kbA);
#pragma unroll
            for (int nj = 0; nj < 4; ++nj)
                frag_b_bf16(bf[nj], &Ws[cb][wn * 64 + nj * 16 + l16][ks * 32 + kbB]);
#pragma unroll
            for (int mi = 0; mi < 2; ++mi)
#pragma unroll
                for (int nj = 0; nj < 4; ++nj)
                    acc[mi][nj] = wmma_bf16(af[mi], bf[nj], acc[mi][nj]);
        }
    }

#pragma unroll
    for (int nj = 0; nj < 4; ++nj) {
        const int col = n0 + wn * 64 + nj * 16 + l16;
        const float bv = bias[col];
#pragma unroll
        for (int mi = 0; mi < 2; ++mi)
#pragma unroll
            for (int r = 0; r < 8; ++r) {
                const int row = m0 + wm * 32 + mi * 16 + r + half * 8;
                store_out(&C[(size_t)row * N + col], (acc[mi][nj][r] + bv) * alpha);
            }
    }
}

// ---------------------------------------------------------------------------
// Flash attention: block = (128 q rows, head, batch); 8 waves x 16 q rows.
// S^T = K Q^T ; O^T = V^T P^T  -> lane owns one q column end-to-end:
// per-lane m/l state, per-lane rescale, in-lane reductions (+1 xor-16 each),
// packed b128 output stores. K/V 32x64 bf16 tiles async-DMA'd, double-buffered.
// Q pre-scaled by (1/sqrt(64))*log2(e) in the Q-projection epilogue.
// ---------------------------------------------------------------------------
#define ATT_SQ   2048
#define ATT_SKV  2048
#define ATT_H    1024
#define ATT_KVH  256
#define ATT_D    64

__global__ __launch_bounds__(256)
void attn_fwd_bf16wmma(const ushort_t* __restrict__ Qp, const ushort_t* __restrict__ Kp,
                       const ushort_t* __restrict__ Vp, ushort_t* __restrict__ Op) {
    __shared__ ushort_t Kt[2][32][72];   // 144B row stride
    __shared__ ushort_t Vt[2][32][72];
    __shared__ ushort_t Pw[8][16][40];   // per-wave P^T scratch [q][kv], 80B stride

    const int tid  = threadIdx.x;
    const int lane = tid & 31;
    const int wave = tid >> 5;
    const int half = lane >> 4;
    const int l16  = lane & 15;
    const int kbA  = half * 8;
    const int kbB  = half * 8;

    const int qb  = blockIdx.x * 128 + wave * 16;
    const int h   = blockIdx.y;
    const int b   = blockIdx.z;
    const int kvh = h >> 2;                       // GROUP_SIZE = 4

    const ushort_t* qbase = Qp + (size_t)b * ATT_SQ  * ATT_H   + (size_t)h   * ATT_D;
    const ushort_t* kbase = Kp + (size_t)b * ATT_SKV * ATT_KVH + (size_t)kvh * ATT_D;
    const ushort_t* vbase = Vp + (size_t)b * ATT_SKV * ATT_KVH + (size_t)kvh * ATT_D;

    // Q^T B-fragments (lane n = q = l16): d slices 0..31 / 32..63
    Frag16 qf0, qf1;
    {
        const ushort_t* qr = qbase + (size_t)(qb + l16) * ATT_H;
        frag_b_bf16(qf0, qr + kbB);
        frag_b_bf16(qf1, qr + 32 + kbB);
    }

    float m_i = -3.0e38f, l_i = 0.0f;    // this lane's q column (q = qb + l16)
    v8f o[4];                            // O^T: tile j = d rows 16j..16j+15
#pragma unroll
    for (int j = 0; j < 4; ++j)
#pragma unroll
        for (int r = 0; r < 8; ++r) o[j][r] = 0.0f;

    // per-wave DMA: wave copies rows 4w..4w+3, lane -> (row, 16B chunk)
    const int crow = wave * 4 + (lane >> 3);
    const int ccol = (lane & 7) * 8;             // shorts
    auto issue = [&](int kv0, int bufi) {
        async_cp16(kbase + (size_t)(kv0 + crow) * ATT_KVH + ccol, &Kt[bufi][crow][ccol]);
        async_cp16(vbase + (size_t)(kv0 + crow) * ATT_KVH + ccol, &Vt[bufi][crow][ccol]);
    };

    const int nslab = ATT_SKV / 32;
    issue(0, 0);
    for (int it = 0; it < nslab; ++it) {
        wait_async0();
        __syncthreads();
        if (it + 1 < nslab) issue((it + 1) * 32, (it + 1) & 1);
        const int cb = it & 1;

        // S^T[32kv x 16q] = K Q^T : two 16-kv-row tiles x two 32-deep d steps
        v8f st[2];
#pragma unroll
        for (int t = 0; t < 2; ++t)
#pragma unroll
            for (int r = 0; r < 8; ++r) st[t][r] = 0.0f;
#pragma unroll
        for (int t = 0; t < 2; ++t) {
            Frag16 ka0, ka1;
            const ushort_t* krow = &Kt[cb][t * 16 + l16][0];
            frag_a_bf16(ka0, krow, kbA);
            frag_a_bf16(ka1, krow + 32, kbA);
            st[t] = wmma_bf16(ka0, qf0, st[t]);
            st[t] = wmma_bf16(ka1, qf1, st[t]);
        }

        // online softmax, in-lane over 16 kv values + one xor-16 combine
        float mt = st[0][0];
#pragma unroll
        for (int r = 1; r < 8; ++r) mt = fmaxf(mt, st[0][r]);
#pragma unroll
        for (int r = 0; r < 8; ++r) mt = fmaxf(mt, st[1][r]);
        mt = fmaxf(mt, __shfl_xor(mt, 16, 32));
        const float newm = fmaxf(m_i, mt);
        const float corr = fast_exp2(m_i - newm);
        m_i = newm;

#pragma unroll
        for (int t = 0; t < 2; ++t)
#pragma unroll
            for (int r = 0; r < 8; ++r)
                st[t][r] = fast_exp2(st[t][r] - m_i);

        float ls = 0.0f;
#pragma unroll
        for (int t = 0; t < 2; ++t)
#pragma unroll
            for (int r = 0; r < 8; ++r) ls += st[t][r];
        ls += __shfl_xor(ls, 16, 32);
        l_i = l_i * corr + ls;

        // rescale O^T: lane owns q -> uniform per-lane multiply, no shuffles
#pragma unroll
        for (int j = 0; j < 4; ++j)
#pragma unroll
            for (int r = 0; r < 8; ++r) o[j][r] *= corr;

        // P^T (C-layout) -> Pw[q][kv] : packed b128 store per (lane, t)
#pragma unroll
        for (int t = 0; t < 2; ++t) {
            Frag16 pk;
#pragma unroll
            for (int r = 0; r < 8; ++r) pk.u[r] = f2bf(st[t][r]);
            *(uint4*)&Pw[wave][l16][half * 8 + t * 16] = ((uint4*)pk.u)[0];
        }
        // P^T as B-fragment (lane n = q, k = kv): contiguous read
        Frag16 pb;
        frag_b_bf16(pb, &Pw[wave][l16][kbB]);

        // O^T[64d x 16q] += V^T[64d x 32kv] P^T[32kv x 16q]
#pragma unroll
        for (int j = 0; j < 4; ++j) {
            Frag16 va;   // A frag of V^T: lane row m = d = 16j+l16, k = kv
#pragma unroll
            for (int i = 0; i < 16; ++i)
                va.u[i] = Vt[cb][kbA + (i & 7) + ((i & 8) << 1)][j * 16 + l16];
            o[j] = wmma_bf16(va, pb, o[j]);
        }
    }

    // normalize in-lane and store O^T rows as packed b128 along the hidden dim
    const float rinv = 1.0f / l_i;
    ushort_t* orow = Op + (size_t)b * ATT_SQ * ATT_H
                        + (size_t)(qb + l16) * ATT_H + h * ATT_D;
#pragma unroll
    for (int j = 0; j < 4; ++j) {
        Frag16 pk;
#pragma unroll
        for (int r = 0; r < 8; ++r) pk.u[r] = f2bf(o[j][r] * rinv);
        *(uint4*)(orow + j * 16 + half * 8) = ((uint4*)pk.u)[0];
    }
}

// ---------------------------------------------------------------------------
// Launch: convert operands -> Q/K/V proj -> attention -> O proj
// ---------------------------------------------------------------------------
extern "C" void kernel_launch(void* const* d_in, const int* in_sizes, int n_in,
                              void* d_out, int out_size, void* d_ws, size_t ws_size,
                              hipStream_t stream) {
    const float* query = (const float*)d_in[0];
    const float* key   = (const float*)d_in[1];
    const float* value = (const float*)d_in[2];
    const float* Wq    = (const float*)d_in[3];
    const float* bq    = (const float*)d_in[4];
    const float* Wk    = (const float*)d_in[5];
    const float* bk    = (const float*)d_in[6];
    const float* Wv    = (const float*)d_in[7];
    const float* bv    = (const float*)d_in[8];
    const float* Wo    = (const float*)d_in[9];
    const float* bo    = (const float*)d_in[10];
    float* out = (float*)d_out;

    const size_t M = 4096;                    // B * SQ
    ushort_t* Wqb = (ushort_t*)d_ws;          // 1024*1024
    ushort_t* Wkb = Wqb + 1024 * 1024;        // 256*1024
    ushort_t* Wvb = Wkb + 256 * 1024;         // 256*1024
    ushort_t* Wob = Wvb + 256 * 1024;         // 1024*1024
    ushort_t* qbf = Wob + 1024 * 1024;        // 4096*1024 (query bf16)
    ushort_t* kbf = qbf + M * 1024;           // 4096*1024 (key bf16)
    ushort_t* vbf = kbf + M * 1024;           // 4096*1024 (value bf16)
    ushort_t* qproj = vbf + M * 1024;         // 4096*1024
    ushort_t* kproj = qproj + M * 1024;       // 4096*256
    ushort_t* vproj = kproj + M * 256;        // 4096*256
    ushort_t* aout  = vproj + M * 256;        // 4096*1024

    dim3 blk(256);
    const int NW = 1024 * 1024, NKV = 256 * 1024, NA = 4096 * 1024;
    cvt_f32_bf16<<<dim3(NW / 2048),  blk, 0, stream>>>(Wq, Wqb, NW);
    cvt_f32_bf16<<<dim3(NKV / 2048), blk, 0, stream>>>(Wk, Wkb, NKV);
    cvt_f32_bf16<<<dim3(NKV / 2048), blk, 0, stream>>>(Wv, Wvb, NKV);
    cvt_f32_bf16<<<dim3(NW / 2048),  blk, 0, stream>>>(Wo, Wob, NW);
    cvt_f32_bf16<<<dim3(NA / 2048),  blk, 0, stream>>>(query, qbf, NA);
    cvt_f32_bf16<<<dim3(NA / 2048),  blk, 0, stream>>>(key,   kbf, NA);
    cvt_f32_bf16<<<dim3(NA / 2048),  blk, 0, stream>>>(value, vbf, NA);

    // fold softmax scale (1/sqrt(64)) and log2(e) into the Q projection
    const float qscale = 0.125f * 1.4426950408889634f;
    gemm_bf16<ushort_t><<<dim3(32, 8), blk, 0, stream>>>(qbf, Wqb, bq, qproj, 4096, 1024, 1024, qscale);
    gemm_bf16<ushort_t><<<dim3(32, 2), blk, 0, stream>>>(kbf, Wkb, bk, kproj, 4096,  256, 1024, 1.0f);
    gemm_bf16<ushort_t><<<dim3(32, 2), blk, 0, stream>>>(vbf, Wvb, bv, vproj, 4096,  256, 1024, 1.0f);

    attn_fwd_bf16wmma<<<dim3(16, 16, 2), blk, 0, stream>>>(qproj, kproj, vproj, aout);

    gemm_bf16<float><<<dim3(32, 8), blk, 0, stream>>>(aout, Wob, bo, out, 4096, 1024, 1024, 1.0f);
}
